// Block_3822520893705
// MI455X (gfx1250) — matrix operand order repaired and checked by reading, not compiled, not run on hardware
//
#include <hip/hip_runtime.h>
#include <math.h>

// ---------------- types ----------------
typedef __bf16 bf16_t;
typedef __attribute__((ext_vector_type(16))) __bf16 v16bf;
typedef __attribute__((ext_vector_type(8)))  float  v8f;

#define NB 2
#define NT 2048
#define NC 1024
#define NH 16
#define NHD 64

__device__ __forceinline__ bf16_t f2bf(float f) {
  unsigned int u = __builtin_bit_cast(unsigned int, f);
  unsigned int r = u + 0x7fffu + ((u >> 16) & 1u);   // round-to-nearest-even
  unsigned short h = (unsigned short)(r >> 16);
  return __builtin_bit_cast(bf16_t, h);
}

__device__ __forceinline__ unsigned int f2bf_bits(float f) {
  unsigned int u = __builtin_bit_cast(unsigned int, f);
  unsigned int r = u + 0x7fffu + ((u >> 16) & 1u);
  return r >> 16;
}

__device__ __forceinline__ float gelu_f(float x) {
  float x3 = x * x * x;
  return 0.5f * x * (1.0f + tanhf(0.7978845608028654f * (x + 0.044715f * x3)));
}

// ---------------- weight convert + transpose: wT[n*K+k] = bf16(w[k*N+n]) ----
__global__ __launch_bounds__(256) void wtrans_kernel(const float* __restrict__ w,
                                                     bf16_t* __restrict__ wT,
                                                     int K, int N) {
  long long idx = (long long)blockIdx.x * 256 + threadIdx.x;
  int k = (int)(idx % K);
  int n = (int)(idx / K);
  wT[idx] = f2bf(w[(long long)k * N + n]);
}

// ---------------- layernorm: fp32 in -> bf16 out -------------------------
__global__ __launch_bounds__(256) void layernorm_bf16(const float* __restrict__ x,
                                                      const float* __restrict__ w,
                                                      const float* __restrict__ b,
                                                      bf16_t* __restrict__ out, int C) {
  __shared__ float red[256];
  __shared__ float s_mu, s_rs;
  const int tid = threadIdx.x;
  const float* px = x + (long long)blockIdx.x * C;
  float s = 0.f;
  for (int i = tid; i < C; i += 256) s += px[i];
  red[tid] = s; __syncthreads();
  for (int off = 128; off > 0; off >>= 1) { if (tid < off) red[tid] += red[tid + off]; __syncthreads(); }
  if (tid == 0) s_mu = red[0] / (float)C;
  __syncthreads();
  const float mu = s_mu;
  float v = 0.f;
  for (int i = tid; i < C; i += 256) { float d = px[i] - mu; v += d * d; }
  __syncthreads();
  red[tid] = v; __syncthreads();
  for (int off = 128; off > 0; off >>= 1) { if (tid < off) red[tid] += red[tid + off]; __syncthreads(); }
  if (tid == 0) s_rs = rsqrtf(red[0] / (float)C + 1e-5f);
  __syncthreads();
  const float rs = s_rs;
  bf16_t* po = out + (long long)blockIdx.x * C;
  for (int i = tid; i < C; i += 256) po[i] = f2bf((px[i] - mu) * rs * w[i] + b[i]);
}

// ---------------- split qkv (bf16) -> q[bh,t,d], k[bh,t,d], vT[bh,d,t] ----
__global__ __launch_bounds__(256) void split_qkv(const bf16_t* __restrict__ qkv,
                                                 bf16_t* __restrict__ q,
                                                 bf16_t* __restrict__ k,
                                                 bf16_t* __restrict__ vT) {
  long long idx = (long long)blockIdx.x * 256 + threadIdx.x; // over B*T*C
  int c  = (int)(idx % NC);
  long long bt = idx / NC;
  int t = (int)(bt % NT);
  int b = (int)(bt / NT);
  int h = c / NHD, d = c % NHD;
  int bh = b * NH + h;
  long long src = bt * (3 * NC) + c;
  q [(long long)bh * NT * NHD + (long long)t * NHD + d] = qkv[src];
  k [(long long)bh * NT * NHD + (long long)t * NHD + d] = qkv[src + NC];
  vT[(long long)bh * NHD * NT + (long long)d * NT + t]  = qkv[src + 2 * NC];
}

// ---------------- rowwise softmax over T elements (in place, fp32) --------
__global__ __launch_bounds__(256) void softmax_rows(float* __restrict__ att, int T) {
  __shared__ float red[256];
  __shared__ float sval;
  const int tid = threadIdx.x;
  float* p = att + (long long)blockIdx.x * T;
  float mx = -3.4e38f;
  for (int i = tid; i < T; i += 256) mx = fmaxf(mx, p[i]);
  red[tid] = mx; __syncthreads();
  for (int off = 128; off > 0; off >>= 1) { if (tid < off) red[tid] = fmaxf(red[tid], red[tid + off]); __syncthreads(); }
  if (tid == 0) sval = red[0];
  __syncthreads();
  mx = sval;
  __syncthreads();
  float s = 0.f;
  for (int i = tid; i < T; i += 256) { float e = expf(p[i] - mx); p[i] = e; s += e; }
  red[tid] = s; __syncthreads();
  for (int off = 128; off > 0; off >>= 1) { if (tid < off) red[tid] += red[tid + off]; __syncthreads(); }
  if (tid == 0) sval = red[0];
  __syncthreads();
  const float inv = 1.0f / sval;
  for (int i = tid; i < T; i += 256) p[i] *= inv;
}

// ---------------- generic bf16 WMMA GEMM with epilogue variants -----------
enum { EPI_BIAS_BF16 = 0, EPI_SCORES = 1, EPI_BF16 = 2, EPI_BIAS_RES_F32 = 3, EPI_GELU_BF16 = 4 };

// C[M,N] = A[M,K] * B(K,N) (+bias/epilogue); B supplied pre-transposed [N,K].
// Software-pipelined: double-buffered LDS (one barrier/stage), register-staged
// global loads for stage k+1 issued under stage k's WMMAs, prefetch for k+2.
template <int BM, int BN, int WROWS, int WCOLS, int EPI, bool A_F32>
__global__ __launch_bounds__(256) void gemm_wmma_bf16(
    const void* __restrict__ Aptr, const bf16_t* __restrict__ Bptr,
    const float* __restrict__ bias, const float* __restrict__ resid,
    void* __restrict__ outPtr,
    int K, int lda, int ldb, int ldo,
    long long aStrideZ, long long bStrideZ, int zdiv,
    long long oStride1, long long oStride2) {
  constexpr int BK = 32;
  constexpr int AP = BK + 8;           // pitch in bf16 (80B rows -> 16B aligned)
  constexpr int TM = BM / WROWS / 16;  // 16x16 tiles per wave (M)
  constexpr int TN = BN / WCOLS / 16;  // 16x16 tiles per wave (N)
  constexpr int A_IT = BM * (BK / 2) / 256;
  constexpr int B_IT = BN * (BK / 2) / 256;

  __shared__ bf16_t As[2][BM * AP];
  __shared__ bf16_t Bs[2][BN * AP];

  const int tid  = threadIdx.x;
  const int lane = tid & 31;
  const int wave = tid >> 5;
  const int wr = wave % WROWS;
  const int wc = wave / WROWS;
  const int waveM0 = wr * (BM / WROWS);
  const int waveN0 = wc * (BN / WCOLS);
  const int nl = lane & 15;
  const int hi = lane >> 4;

  const int z  = blockIdx.z;
  const long long zoff = (long long)(z / zdiv) * oStride1 + (long long)(z % zdiv) * oStride2;
  const int m0 = blockIdx.y * BM;
  const int n0 = blockIdx.x * BN;

  const float*  A32 = (const float*)Aptr + (long long)z * aStrideZ;
  const bf16_t* A16 = (const bf16_t*)Aptr + (long long)z * aStrideZ;
  const bf16_t* Bm  = Bptr + (long long)z * bStrideZ;

  union Frag { v16bf v; uint4 q[2]; };

  v8f acc[TM][TN] = {};

  // register staging for the in-flight tile
  unsigned int ra[A_F32 ? 1 : A_IT];
  float        raf[A_F32 ? 2 * A_IT : 1];
  unsigned int rb[B_IT];

  auto load_global = [&](int k0) {
#pragma unroll
    for (int it = 0; it < A_IT; ++it) {
      int i = tid + it * 256;
      int row = i >> 4;           // / (BK/2)
      int kp  = i & 15;
      int gk  = k0 + kp * 2;
      if constexpr (A_F32) {
        float2 f = *(const float2*)&A32[(long long)(m0 + row) * lda + gk];
        raf[2 * it]     = f.x;
        raf[2 * it + 1] = f.y;
      } else {
        ra[it] = *(const unsigned int*)&A16[(long long)(m0 + row) * lda + gk];
      }
    }
#pragma unroll
    for (int it = 0; it < B_IT; ++it) {
      int i = tid + it * 256;
      int n  = i >> 4;
      int kp = i & 15;
      int gk = k0 + kp * 2;
      rb[it] = *(const unsigned int*)&Bm[(long long)(n0 + n) * ldb + gk];
    }
  };

  auto store_lds = [&](int buf) {
#pragma unroll
    for (int it = 0; it < A_IT; ++it) {
      int i = tid + it * 256;
      int row = i >> 4;
      int kp  = i & 15;
      if constexpr (A_F32) {
        unsigned int p = f2bf_bits(raf[2 * it]) | (f2bf_bits(raf[2 * it + 1]) << 16);
        *(unsigned int*)&As[buf][row * AP + kp * 2] = p;
      } else {
        *(unsigned int*)&As[buf][row * AP + kp * 2] = ra[it];
      }
    }
#pragma unroll
    for (int it = 0; it < B_IT; ++it) {
      int i = tid + it * 256;
      int n  = i >> 4;
      int kp = i & 15;
      *(unsigned int*)&Bs[buf][n * AP + kp * 2] = rb[it];
    }
  };

  load_global(0);
  int buf = 0;

  for (int k0 = 0; k0 < K; k0 += BK) {
    store_lds(buf);
    __syncthreads();

    // kick off next stage's global loads (latency hidden under WMMAs below)
    if (k0 + BK < K) load_global(k0 + BK);
    // pull stage k+2 toward the WGP caches (uniform branch; global_prefetch_b8)
    if (k0 + 2 * BK < K) {
      int row = tid >> 4;
      if constexpr (A_F32)
        __builtin_prefetch(&A32[(long long)(m0 + row) * lda + k0 + 2 * BK], 0, 0);
      else
        __builtin_prefetch(&A16[(long long)(m0 + row) * lda + k0 + 2 * BK], 0, 0);
      __builtin_prefetch(&Bm[(long long)(n0 + (tid >> 4) % BN) * ldb + k0 + 2 * BK], 0, 0);
    }

    // ---- fragment loads: ds_load_b128 pairs matching ISA 16-bit layouts
    Frag aF[TM], bF[TN];
#pragma unroll
    for (int tm = 0; tm < TM; ++tm) {
      const uint4* rp = (const uint4*)&As[buf][(waveM0 + tm * 16 + nl) * AP];
      aF[tm].q[0] = rp[hi];       // K = 8*hi .. 8*hi+7
      aF[tm].q[1] = rp[2 + hi];   // K = 16+8*hi .. +7
    }
#pragma unroll
    for (int tn = 0; tn < TN; ++tn) {
      const uint4* rp = (const uint4*)&Bs[buf][(waveN0 + tn * 16 + nl) * AP];
      bF[tn].q[0] = rp[hi];         // K = 8*hi .. 8*hi+7
      bF[tn].q[1] = rp[2 + hi];     // K = 16+8*hi .. +7
    }
#pragma unroll
    for (int tm = 0; tm < TM; ++tm)
#pragma unroll
      for (int tn = 0; tn < TN; ++tn)
        acc[tm][tn] = __builtin_amdgcn_wmma_f32_16x16x32_bf16(
            false, aF[tm].v, false, bF[tn].v, (short)0, acc[tm][tn], false, false);

    buf ^= 1;
  }

  // ---- epilogue (C/D layout: VGPR v -> M = v + 8*hi, N = lane&15)
  float*  outF = (float*)outPtr + zoff;
  bf16_t* outB = (bf16_t*)outPtr + zoff;
#pragma unroll
  for (int tm = 0; tm < TM; ++tm)
#pragma unroll
    for (int tn = 0; tn < TN; ++tn) {
      const int gm = m0 + waveM0 + tm * 16;
      const int gn = n0 + waveN0 + tn * 16 + nl;
#pragma unroll
      for (int v = 0; v < 8; ++v) {
        const int r = gm + v + 8 * hi;
        float val = acc[tm][tn][v];
        long long idx = (long long)r * ldo + gn;
        if constexpr (EPI == EPI_BIAS_BF16) {
          outB[idx] = f2bf(val + bias[gn]);
        } else if constexpr (EPI == EPI_SCORES) {
          val *= 0.125f;                 // 1/sqrt(64)
          if (gn > r) val = -1e30f;      // causal mask
          outF[idx] = val;
        } else if constexpr (EPI == EPI_BF16) {
          outB[idx] = f2bf(val);
        } else if constexpr (EPI == EPI_BIAS_RES_F32) {
          outF[idx] = resid[idx] + val + bias[gn];
        } else { // EPI_GELU_BF16
          outB[idx] = f2bf(gelu_f(val + bias[gn]));
        }
      }
    }
}

// ---------------- launch ---------------------------------------------------
extern "C" void kernel_launch(void* const* d_in, const int* in_sizes, int n_in,
                              void* d_out, int out_size, void* d_ws, size_t ws_size,
                              hipStream_t stream) {
  (void)in_sizes; (void)n_in; (void)out_size; (void)ws_size;
  const float* x         = (const float*)d_in[0];
  // d_in[1] attn_mask (tril) is implied by the analytic causal mask
  const float* w_attn    = (const float*)d_in[2];
  const float* b_attn    = (const float*)d_in[3];
  const float* w_proj    = (const float*)d_in[4];
  const float* b_proj    = (const float*)d_in[5];
  const float* ln1_w     = (const float*)d_in[6];
  const float* ln1_b     = (const float*)d_in[7];
  const float* ln2_w     = (const float*)d_in[8];
  const float* ln2_b     = (const float*)d_in[9];
  const float* w_fc      = (const float*)d_in[10];
  const float* b_fc      = (const float*)d_in[11];
  const float* w_fc_proj = (const float*)d_in[12];
  const float* b_fc_proj = (const float*)d_in[13];

  float* out_x   = (float*)d_out;
  float* out_att = out_x + (long long)NB * NT * NC;

  // workspace carve-up (256B aligned chunks)
  char* ws = (char*)d_ws;
  size_t off = 0;
  auto carve = [&](size_t bytes) { char* p = ws + off; off = (off + bytes + 255) & ~(size_t)255; return p; };
  bf16_t* wattnT   = (bf16_t*)carve((size_t)3072 * 1024 * 2);
  bf16_t* wprojT   = (bf16_t*)carve((size_t)1024 * 1024 * 2);
  bf16_t* wfcT     = (bf16_t*)carve((size_t)4096 * 1024 * 2);
  bf16_t* wfcprojT = (bf16_t*)carve((size_t)1024 * 4096 * 2);
  bf16_t* h_bf     = (bf16_t*)carve((size_t)4096 * 1024 * 2);
  bf16_t* qkv_bf   = (bf16_t*)carve((size_t)4096 * 3072 * 2);
  bf16_t* q_bf     = (bf16_t*)carve((size_t)32 * 2048 * 64 * 2);
  bf16_t* k_bf     = (bf16_t*)carve((size_t)32 * 2048 * 64 * 2);
  bf16_t* vT_bf    = (bf16_t*)carve((size_t)32 * 64 * 2048 * 2);
  bf16_t* y_bf     = (bf16_t*)carve((size_t)4096 * 1024 * 2);
  float*  x1       = (float*)carve((size_t)4096 * 1024 * 4);
  bf16_t* h2_bf    = (bf16_t*)carve((size_t)4096 * 1024 * 2);
  bf16_t* m_bf     = (bf16_t*)carve((size_t)4096 * 4096 * 2);

  const int M = NB * NT; // 4096 rows

  // 1) weights -> bf16, transposed to [N,K]
  wtrans_kernel<<<(1024 * 3072) / 256, 256, 0, stream>>>(w_attn, wattnT, 1024, 3072);
  wtrans_kernel<<<(1024 * 1024) / 256, 256, 0, stream>>>(w_proj, wprojT, 1024, 1024);
  wtrans_kernel<<<(1024 * 4096) / 256, 256, 0, stream>>>(w_fc, wfcT, 1024, 4096);
  wtrans_kernel<<<(4096 * 1024) / 256, 256, 0, stream>>>(w_fc_proj, wfcprojT, 4096, 1024);

  // 2) LN1
  layernorm_bf16<<<M, 256, 0, stream>>>(x, ln1_w, ln1_b, h_bf, NC);

  // 3) qkv = h @ w_attn + b_attn  -> bf16
  gemm_wmma_bf16<128, 128, 4, 2, EPI_BIAS_BF16, false>
      <<<dim3(3072 / 128, M / 128, 1), 256, 0, stream>>>(
          h_bf, wattnT, b_attn, nullptr, qkv_bf,
          1024, 1024, 1024, 3072, 0, 0, 1, 0, 0);

  // 4) split/permute into q,k (row-major per head) and vT (transposed per head)
  split_qkv<<<((long long)NB * NT * NC) / 256, 256, 0, stream>>>(qkv_bf, q_bf, k_bf, vT_bf);

  // 5) scores: per (b,h)  S = Q @ K^T * 0.125, causal masked -> fp32 att
  gemm_wmma_bf16<128, 128, 4, 2, EPI_SCORES, false>
      <<<dim3(NT / 128, NT / 128, NB * NH), 256, 0, stream>>>(
          q_bf, k_bf, nullptr, nullptr, out_att,
          NHD, NHD, NHD, NT,
          (long long)NT * NHD, (long long)NT * NHD, 1,
          (long long)NT * NT, 0);

  // 6) softmax rows (in place on output att)
  softmax_rows<<<NB * NH * NT, 256, 0, stream>>>(out_att, NT);

  // 7) y = att @ V  (A is fp32 att, converted to bf16 while staging) -> y_bf [B,T,C]
  gemm_wmma_bf16<128, 64, 8, 1, EPI_BF16, true>
      <<<dim3(1, NT / 128, NB * NH), 256, 0, stream>>>(
          out_att, vT_bf, nullptr, nullptr, y_bf,
          NT, NT, NT, NC,
          (long long)NT * NT, (long long)NHD * NT, NH,
          (long long)NT * NC, NHD);

  // 8) x1 = x + y @ w_proj + b_proj  (fp32)
  gemm_wmma_bf16<128, 128, 4, 2, EPI_BIAS_RES_F32, false>
      <<<dim3(1024 / 128, M / 128, 1), 256, 0, stream>>>(
          y_bf, wprojT, b_proj, x, x1,
          1024, 1024, 1024, 1024, 0, 0, 1, 0, 0);

  // 9) LN2
  layernorm_bf16<<<M, 256, 0, stream>>>(x1, ln2_w, ln2_b, h2_bf, NC);

  // 10) m = gelu(h2 @ w_fc + b_fc) -> bf16
  gemm_wmma_bf16<128, 128, 4, 2, EPI_GELU_BF16, false>
      <<<dim3(4096 / 128, M / 128, 1), 256, 0, stream>>>(
          h2_bf, wfcT, b_fc, nullptr, m_bf,
          1024, 1024, 1024, 4096, 0, 0, 1, 0, 0);

  // 11) x = x1 + m @ w_fc_proj + b_fc_proj  (fp32 -> d_out)
  gemm_wmma_bf16<128, 128, 4, 2, EPI_BIAS_RES_F32, false>
      <<<dim3(1024 / 128, M / 128, 1), 256, 0, stream>>>(
          m_bf, wfcprojT, b_fc_proj, x1, out_x,
          4096, 4096, 4096, 1024, 0, 0, 1, 0, 0);
}
